// FluxPropLayer_38285338477252
// MI455X (gfx1250) — matrix operand order repaired and checked by reading, not compiled, not run on hardware
//
#include <hip/hip_runtime.h>
#include <hip/hip_bf16.h>
#include <math.h>

#define B_SZ 8192
#define D_SZ 2048
#define H_SZ 2048
#define R_SZ 64
#define K1   (H_SZ + D_SZ)   // 4096 fused K for the liquid GEMM

typedef __bf16 bf16;
typedef __attribute__((ext_vector_type(16))) __bf16 v16bf;
typedef __attribute__((ext_vector_type(8)))  __bf16 v8bf;
typedef __attribute__((ext_vector_type(4)))  __bf16 v4bf;
typedef __attribute__((ext_vector_type(8)))  float  v8f;
typedef __attribute__((ext_vector_type(4)))  float  v4f;
typedef __attribute__((ext_vector_type(4)))  unsigned int u32x4;
typedef __attribute__((ext_vector_type(8)))  int    i32x8;
typedef __attribute__((ext_vector_type(4)))  int    i32x4;

union Frag16 { v16bf v; v8bf h[2]; };

__device__ __forceinline__ bf16 to_bf16(float f) { return (bf16)f; }

// ---------------------------------------------------------------------------
// TDM: DMA one 128(N) x 32(K) bf16 weight tile from global into LDS.
// D# per CDNA5 ISA 8.3/8.4: 2D tensor [2048, 4096] bf16, row stride 4096,
// tile_dim0=32 (contiguous K), tile_dim1=128 (N rows), data_size=2B, type=2.
// 6-arg builtin form (clang-23 / therock-10.0 headers).
// ---------------------------------------------------------------------------
__device__ __forceinline__ void tdm_load_tile(unsigned lds_addr, const bf16* gptr) {
  unsigned long long ga = (unsigned long long)(uintptr_t)gptr;
  u32x4 g0;
  g0[0] = 1u;                                               // count=1, no gather
  g0[1] = lds_addr;                                         // LDS byte address
  g0[2] = (unsigned)(ga & 0xffffffffu);                     // global_addr[31:0]
  g0[3] = (unsigned)((ga >> 32) & 0x01ffffffu) | (2u << 30);// addr[56:32] | type=2
  i32x8 g1;
  g1[0] = 0x10000;                 // workgroup_mask=0, data_size=1 (2 bytes)
  g1[1] = (int)(4096u << 16);      // tensor_dim0 = 4096 (low 16 bits)
  g1[2] = (int)(2048u << 16);      // tensor_dim0 hi=0 | tensor_dim1 = 2048 lo
  g1[3] = (int)(32u << 16);        // tensor_dim1 hi=0 | tile_dim0 = 32
  g1[4] = 128;                     // tile_dim1 = 128, tile_dim2 = 0
  g1[5] = 4096;                    // tensor_dim0_stride = 4096
  g1[6] = 0;
  g1[7] = 0;
  i32x4 z4 = {0, 0, 0, 0};
  i32x8 z8 = {0, 0, 0, 0, 0, 0, 0, 0};
  __builtin_amdgcn_tensor_load_to_lds(g0, g1, z4, z4, z8, 0);
}

// ---------------------------------------------------------------------------
// Pack kernels (weights only; activations are converted in-register).
// ---------------------------------------------------------------------------
__global__ __launch_bounds__(256) void pack_w(const float* __restrict__ W_w,
                                              const float* __restrict__ U_w,
                                              bf16* __restrict__ Wc_bf) {
  long i   = (long)blockIdx.x * blockDim.x + threadIdx.x;
  long idx = i * 4;
  long n   = idx >> 12;
  int  c   = (int)(idx & 4095);
  const float* src = (c < H_SZ) ? (W_w + n * H_SZ + c)
                                : (U_w + n * D_SZ + (c - H_SZ));
  v4f f = *(const v4f*)src;
  v4bf o; o.x = to_bf16(f.x); o.y = to_bf16(f.y); o.z = to_bf16(f.z); o.w = to_bf16(f.w);
  *(v4bf*)(Wc_bf + idx) = o;
}

__global__ __launch_bounds__(256) void pack_misc(const float* __restrict__ WQ,
                                                 const float* __restrict__ WK,
                                                 const float* __restrict__ WV,
                                                 const float* __restrict__ WO,
                                                 const float* __restrict__ alpha_raw,
                                                 bf16* __restrict__ Wqkv_bf,
                                                 bf16* __restrict__ WO_bf,
                                                 float* __restrict__ alpha_s) {
  const int NQKV = 3 * R_SZ * H_SZ;     // 393216
  const int NWO  = H_SZ * R_SZ;         // 131072
  long i = (long)blockIdx.x * blockDim.x + threadIdx.x;
  if (i < NQKV) {
    int r = (int)(i / H_SZ), c = (int)(i % H_SZ);
    float v = (r < 64) ? WQ[r * H_SZ + c]
            : (r < 128) ? WK[(r - 64) * H_SZ + c]
                        : WV[(r - 128) * H_SZ + c];
    Wqkv_bf[i] = to_bf16(v);
  } else if (i < NQKV + NWO) {
    long j = i - NQKV;
    WO_bf[j] = to_bf16(WO[j]);
  } else if (i < NQKV + NWO + H_SZ) {
    long j = i - NQKV - NWO;
    alpha_s[j] = 1.0f / (1.0f + __expf(-alpha_raw[j]));
  }
}

// ---------------------------------------------------------------------------
// GEMM 1 (fused liquid update): C = [h_prev|x_t] @ Wc_bf[2048,4096]^T
//  - A: fp32 loaded directly, converted in-register to bf16 fragments.
//  - B: TDM tensor_load_to_lds double-buffered 128x32 tiles, ds_load frags.
//  - epilogue: h = (1-alpha)*h_prev + tanh(C + U_b); fp32 + bf16 stores.
// Block 128x128, 8 waves (4 M x 2 N), wave tile 32x64 -> 2x4 WMMA accums.
// ---------------------------------------------------------------------------
__global__ __launch_bounds__(256) void gemm_liquid(const float* __restrict__ h_prev,
                                                   const float* __restrict__ x_t,
                                                   const bf16* __restrict__ Wc,
                                                   const float* __restrict__ U_b,
                                                   const float* __restrict__ alpha_s,
                                                   float* __restrict__ h_out,
                                                   bf16* __restrict__ h_bf) {
  __shared__ __align__(16) bf16 ldsB[2][128 * 32];   // 2 x 8KB double buffer

  const int lane = threadIdx.x & 31;
  const int wave = threadIdx.x >> 5;
  const bool w0  = (wave == 0);
  const int wm = wave & 3, wn = wave >> 2;
  const int m0 = blockIdx.y * 128 + wm * 32;
  const int n0 = blockIdx.x * 128;
  const int half = lane >> 4, l16 = lane & 15;

  const unsigned ldsAddr[2] = {
      (unsigned)(uintptr_t)(void*)&ldsB[0][0],
      (unsigned)(uintptr_t)(void*)&ldsB[1][0]};

  v8f acc[2][4] = {};
  const float* ArowH[2];
  const float* ArowX[2];
#pragma unroll
  for (int mt = 0; mt < 2; ++mt) {
    ArowH[mt] = h_prev + (long)(m0 + mt * 16 + l16) * H_SZ;
    ArowX[mt] = x_t    + (long)(m0 + mt * 16 + l16) * D_SZ;
  }
  const bf16* WcBase = Wc + (long)n0 * K1;

  if (w0) tdm_load_tile(ldsAddr[0], WcBase);

  // One 32-wide K step against a statically-indexed LDS buffer.
#define GEMM_STEP(BUF, KK)                                                     \
  {                                                                            \
    const int kk = (KK);                                                       \
    __builtin_amdgcn_s_wait_tensorcnt(0);  /* per-wave; free for waves 1..7 */ \
    __syncthreads();                       /* publish buffer BUF */            \
    if (w0 && (kk + 32 < K1))              /* DMA next tile into other buf */  \
      tdm_load_tile(ldsAddr[(BUF) ^ 1], WcBase + kk + 32);                     \
    v8f af[2][2];                          /* independent fp32 staging */      \
    _Pragma("unroll")                                                          \
    for (int mt = 0; mt < 2; ++mt) {                                           \
      const float* src = (kk < H_SZ) ? (ArowH[mt] + kk)                        \
                                     : (ArowX[mt] + (kk - H_SZ));              \
      af[mt][0] = *(const v8f*)(src + half * 8);                               \
      af[mt][1] = *(const v8f*)(src + 16 + half * 8);                          \
    }                                                                          \
    Frag16 b[4];                                                               \
    _Pragma("unroll")                                                          \
    for (int nt = 0; nt < 4; ++nt) {                                           \
      const bf16* lb = &ldsB[BUF][(wn * 64 + nt * 16 + l16) * 32 + half * 8]; \
      b[nt].h[0] = *(const v8bf*)(lb);                                         \
      b[nt].h[1] = *(const v8bf*)(lb + 16);                                    \
    }                                                                          \
    Frag16 a[2];                                                               \
    _Pragma("unroll")                                                          \
    for (int mt = 0; mt < 2; ++mt)                                             \
      _Pragma("unroll")                                                        \
      for (int e = 0; e < 8; ++e) {                                            \
        a[mt].v[e]     = to_bf16(af[mt][0][e]);                                \
        a[mt].v[e + 8] = to_bf16(af[mt][1][e]);                                \
      }                                                                        \
    _Pragma("unroll")                                                          \
    for (int mt = 0; mt < 2; ++mt)                                             \
      _Pragma("unroll")                                                        \
      for (int nt = 0; nt < 4; ++nt)                                           \
        acc[mt][nt] = __builtin_amdgcn_wmma_f32_16x16x32_bf16(                 \
            false, a[mt].v, false, b[nt].v, (short)0, acc[mt][nt],             \
            false, false);                                                     \
  }

  for (int k0 = 0; k0 < K1; k0 += 64) {   // 2 steps per iter, static buffers
    GEMM_STEP(0, k0);
    GEMM_STEP(1, k0 + 32);
  }
#undef GEMM_STEP

#pragma unroll
  for (int mt = 0; mt < 2; ++mt) {
#pragma unroll
    for (int nt = 0; nt < 4; ++nt) {
      const int col = n0 + (wn * 64) + nt * 16 + l16;
      const float al = alpha_s[col];
      const float bias = U_b[col];
#pragma unroll
      for (int r = 0; r < 8; ++r) {
        const int row = m0 + mt * 16 + r + half * 8;
        const long off = (long)row * H_SZ + col;
        float st = tanhf(acc[mt][nt][r] + bias);
        float h  = (1.0f - al) * h_prev[off] + st;
        h_out[off] = h;
        h_bf[off]  = to_bf16(h);
      }
    }
  }
}

// ---------------------------------------------------------------------------
// GEMM 2: qkv[B,192] = h_bf[B,2048] @ Wqkv_bf[192,2048]^T   (fp32 out)
// Block 128x64, wave tile 32x32 -> 2x2 accums.
// ---------------------------------------------------------------------------
__global__ __launch_bounds__(256) void gemm_qkv(const bf16* __restrict__ A,
                                                const bf16* __restrict__ Wqkv,
                                                float* __restrict__ qkv) {
  const int lane = threadIdx.x & 31;
  const int wave = threadIdx.x >> 5;
  const int wm = wave & 3, wn = wave >> 2;
  const int m0 = blockIdx.y * 128 + wm * 32;
  const int n0 = blockIdx.x * 64 + wn * 32;
  const int half = lane >> 4, l16 = lane & 15;

  v8f acc[2][2] = {};
  const bf16* Arow[2];
  const bf16* Brow[2];
#pragma unroll
  for (int mt = 0; mt < 2; ++mt) Arow[mt] = A    + (long)(m0 + mt * 16 + l16) * H_SZ;
#pragma unroll
  for (int nt = 0; nt < 2; ++nt) Brow[nt] = Wqkv + (long)(n0 + nt * 16 + l16) * H_SZ;

  for (int k0 = 0; k0 < H_SZ; k0 += 32) {
    Frag16 a[2], b[2];
#pragma unroll
    for (int mt = 0; mt < 2; ++mt) {
      a[mt].h[0] = *(const v8bf*)(Arow[mt] + k0 + half * 8);
      a[mt].h[1] = *(const v8bf*)(Arow[mt] + k0 + 16 + half * 8);
    }
#pragma unroll
    for (int nt = 0; nt < 2; ++nt) {
      b[nt].h[0] = *(const v8bf*)(Brow[nt] + k0 + half * 8);
      b[nt].h[1] = *(const v8bf*)(Brow[nt] + k0 + 16 + half * 8);
    }
#pragma unroll
    for (int mt = 0; mt < 2; ++mt)
#pragma unroll
      for (int nt = 0; nt < 2; ++nt)
        acc[mt][nt] = __builtin_amdgcn_wmma_f32_16x16x32_bf16(
            false, a[mt].v, false, b[nt].v, (short)0, acc[mt][nt], false, false);
  }

#pragma unroll
  for (int mt = 0; mt < 2; ++mt)
#pragma unroll
    for (int nt = 0; nt < 2; ++nt) {
      const int col = n0 + nt * 16 + l16;
#pragma unroll
      for (int r = 0; r < 8; ++r) {
        const int row = m0 + mt * 16 + r + half * 8;
        qkv[(long)row * 192 + col] = acc[mt][nt][r];
      }
    }
}

// ---------------------------------------------------------------------------
// Routing: scores are rank-1 (q_i * k_j / 8) => per-i scalar softmax over k,v.
// One wave32 per batch row; lane holds elements j and j+32. Never builds [R,R].
// ---------------------------------------------------------------------------
__global__ __launch_bounds__(256) void route_softmax(const float* __restrict__ qkv,
                                                     bf16* __restrict__ routed_bf) {
  const int lane = threadIdx.x & 31;
  const int wave = threadIdx.x >> 5;
  const long row = (long)blockIdx.x * 8 + wave;
  const float* base = qkv + row * 192;
  const float q0 = base[lane],       q1 = base[lane + 32];
  const float k0 = base[64 + lane],  k1 = base[96 + lane];
  const float v0 = base[128 + lane], v1 = base[160 + lane];

  float kmx = fmaxf(k0, k1), kmn = fminf(k0, k1);
#pragma unroll
  for (int off = 16; off > 0; off >>= 1) {
    kmx = fmaxf(kmx, __shfl_xor(kmx, off, 32));
    kmn = fminf(kmn, __shfl_xor(kmn, off, 32));
  }

  float r0 = 0.f, r1 = 0.f;
  for (int i = 0; i < 64; ++i) {
    float qi = (i < 32) ? __shfl(q0, i, 32) : __shfl(q1, i - 32, 32);
    float s  = qi * 0.125f;                         // 1/sqrt(R)
    float m  = fmaxf(s * kmx, s * kmn);             // exact row max (monotone in k)
    float e0 = __expf(s * k0 - m);
    float e1 = __expf(s * k1 - m);
    float num = e0 * v0 + e1 * v1;
    float den = e0 + e1;
#pragma unroll
    for (int off = 16; off > 0; off >>= 1) {
      num += __shfl_xor(num, off, 32);
      den += __shfl_xor(den, off, 32);
    }
    float val = num / den;
    if (lane == (i & 31)) { if (i < 32) r0 = val; else r1 = val; }
  }
  routed_bf[row * 64 + lane]      = to_bf16(r0);
  routed_bf[row * 64 + lane + 32] = to_bf16(r1);
}

// ---------------------------------------------------------------------------
// GEMM 3 (fused output): y = gamma * (routed_bf[B,64] @ WO_bf[2048,64]^T) + x_t
// Block 128x128, wave tile 32x64 -> 2x4 accums, K loop = 2 WMMA steps.
// ---------------------------------------------------------------------------
__global__ __launch_bounds__(256) void gemm_out(const bf16* __restrict__ Rv,
                                                const bf16* __restrict__ WOb,
                                                const float* __restrict__ x_t,
                                                const float* __restrict__ gamma,
                                                float* __restrict__ y_out) {
  const int lane = threadIdx.x & 31;
  const int wave = threadIdx.x >> 5;
  const int wm = wave & 3, wn = wave >> 2;
  const int m0 = blockIdx.y * 128 + wm * 32;
  const int n0 = blockIdx.x * 128 + wn * 64;
  const int half = lane >> 4, l16 = lane & 15;
  const float g = gamma[0];

  v8f acc[2][4] = {};
  const bf16* Arow[2];
  const bf16* Brow[4];
#pragma unroll
  for (int mt = 0; mt < 2; ++mt) Arow[mt] = Rv  + (long)(m0 + mt * 16 + l16) * R_SZ;
#pragma unroll
  for (int nt = 0; nt < 4; ++nt) Brow[nt] = WOb + (long)(n0 + nt * 16 + l16) * R_SZ;

#pragma unroll
  for (int k0 = 0; k0 < R_SZ; k0 += 32) {
    Frag16 a[2], b[4];
#pragma unroll
    for (int mt = 0; mt < 2; ++mt) {
      a[mt].h[0] = *(const v8bf*)(Arow[mt] + k0 + half * 8);
      a[mt].h[1] = *(const v8bf*)(Arow[mt] + k0 + 16 + half * 8);
    }
#pragma unroll
    for (int nt = 0; nt < 4; ++nt) {
      b[nt].h[0] = *(const v8bf*)(Brow[nt] + k0 + half * 8);
      b[nt].h[1] = *(const v8bf*)(Brow[nt] + k0 + 16 + half * 8);
    }
#pragma unroll
    for (int mt = 0; mt < 2; ++mt)
#pragma unroll
      for (int nt = 0; nt < 4; ++nt)
        acc[mt][nt] = __builtin_amdgcn_wmma_f32_16x16x32_bf16(
            false, a[mt].v, false, b[nt].v, (short)0, acc[mt][nt], false, false);
  }

#pragma unroll
  for (int mt = 0; mt < 2; ++mt)
#pragma unroll
    for (int nt = 0; nt < 4; ++nt) {
      const int col = n0 + nt * 16 + l16;
#pragma unroll
      for (int r = 0; r < 8; ++r) {
        const int row = m0 + mt * 16 + r + half * 8;
        const long off = (long)row * H_SZ + col;
        y_out[off] = g * acc[mt][nt][r] + x_t[off];
      }
    }
}

// ---------------------------------------------------------------------------
extern "C" void kernel_launch(void* const* d_in, const int* in_sizes, int n_in,
                              void* d_out, int out_size, void* d_ws, size_t ws_size,
                              hipStream_t stream) {
  const float* h_prev    = (const float*)d_in[0];
  const float* x_t       = (const float*)d_in[1];
  const float* U_w       = (const float*)d_in[2];
  const float* U_b       = (const float*)d_in[3];
  const float* W_w       = (const float*)d_in[4];
  const float* alpha_raw = (const float*)d_in[5];
  const float* WQ        = (const float*)d_in[6];
  const float* WK        = (const float*)d_in[7];
  const float* WV        = (const float*)d_in[8];
  const float* WO        = (const float*)d_in[9];
  const float* gamma     = (const float*)d_in[10];

  float* h_out = (float*)d_out;                     // [B,H] h_t
  float* y_out = (float*)d_out + (long)B_SZ * H_SZ; // [B,H] y_t

  // workspace layout (bytes): ~56 MB total
  char* ws = (char*)d_ws;
  bf16*  Wc_bf     = (bf16*)(ws);                     // 16 MB  [2048,4096]
  bf16*  h_bf      = (bf16*)(ws + 16777216);          // 32 MB  [B,2048]
  float* qkv       = (float*)(ws + 50331648);         //  6 MB  [B,192]
  bf16*  routed_bf = (bf16*)(ws + 56623104);          //  1 MB  [B,64]
  bf16*  Wqkv_bf   = (bf16*)(ws + 57671680);          // .75 MB [192,2048]
  bf16*  WO_bf     = (bf16*)(ws + 58458112);          // .25 MB [2048,64]
  float* alpha_s   = (float*)(ws + 58720256);         //  8 KB  [2048]

  // 1) pack/convert weights (activations converted in-register in GEMM1)
  pack_w   <<<8192, 256, 0, stream>>>(W_w, U_w, Wc_bf);
  pack_misc<<<2056, 256, 0, stream>>>(WQ, WK, WV, WO, alpha_raw,
                                      Wqkv_bf, WO_bf, alpha_s);
  // 2) liquid state GEMM (TDM-staged weights) + fused tanh/alpha-gate epilogue
  gemm_liquid<<<dim3(H_SZ / 128, B_SZ / 128), 256, 0, stream>>>(
      h_prev, x_t, Wc_bf, U_b, alpha_s, h_out, h_bf);
  // 3) QKV projection
  gemm_qkv<<<dim3(3, B_SZ / 128), 256, 0, stream>>>(h_bf, Wqkv_bf, qkv);
  // 4) rank-1 softmax routing (one wave per row)
  route_softmax<<<B_SZ / 8, 256, 0, stream>>>(qkv, routed_bf);
  // 5) output projection + fused gamma*h_tilde + x residual
  gemm_out<<<dim3(H_SZ / 128, B_SZ / 128), 256, 0, stream>>>(
      routed_bf, WO_bf, x_t, gamma, y_out);
}
